// ConvEnc_72799695667608
// MI455X (gfx1250) — matrix operand order repaired and checked by reading, not compiled, not run on hardware
//
#include <hip/hip_runtime.h>

typedef __attribute__((ext_vector_type(16))) _Float16 v16h;
typedef __attribute__((ext_vector_type(8)))  float    v8f;

#define C_OUT 128
#define HH 128
#define WW 128
#define NB 32
#define BN_EPS 1e-5f

// One workgroup per (n, hrow): 8 waves x (16 channels each) x 8 pixel tiles of 16.
// GEMM view: out[128 ch, 128 pix] = Wmat[128,9] * im2col[9,128], K padded to 32
// for v_wmma_f32_16x16x32_f16.  BN folded to scale/bias; LIF firing-rate epilogue
// runs as ONE dynamic-T loop per tile with a branch-free 8-wide vector body.
// Output (268 MB, write-once, > L2) is streamed with non-temporal stores.
__global__ __launch_bounds__(256)
void ConvEnc_lif_wmma_kernel(const float* __restrict__ x,
                             const float* __restrict__ wgt,
                             const float* __restrict__ gamma,
                             const float* __restrict__ beta,
                             const float* __restrict__ rmean,
                             const float* __restrict__ rvar,
                             const int*   __restrict__ Tptr,
                             const float* __restrict__ tauptr,
                             float* __restrict__ out) {
    __shared__ _Float16 srow[3][132];   // 3 input rows, cols shifted +1 (halo), padded
    __shared__ float s_inv[C_OUT];
    __shared__ float s_bias[C_OUT];

    const int n    = blockIdx.y;
    const int hrow = blockIdx.x;
    const int tid  = threadIdx.x;

    // ---- stage input halo rows (h-1, h, h+1) into LDS as f16, zero-padded ----
    for (int idx = tid; idx < 3 * 130; idx += 256) {
        int j   = idx / 130;           // 0..2 -> row h-1+j
        int col = idx % 130;           // srow col; global x = col-1
        int gy  = hrow - 1 + j;
        int gx  = col - 1;
        float v = 0.0f;
        if (gy >= 0 && gy < HH && gx >= 0 && gx < WW)
            v = x[((size_t)n * HH + gy) * WW + gx];
        srow[j][col] = (_Float16)v;
    }
    // ---- fold BN into per-channel scale/bias ----
    if (tid < C_OUT) {
        float inv = gamma[tid] * rsqrtf(rvar[tid] + BN_EPS);
        s_inv[tid]  = inv;
        s_bias[tid] = beta[tid] - rmean[tid] * inv;
    }
    __syncthreads();

    const int wave = tid >> 5;       // 0..7 -> channel block
    const int lane = tid & 31;
    const int hi   = lane >> 4;      // half-wave selector (ISA 16-bit operand layout)
    const int mr   = lane & 15;

    // ---- A fragment: 16x32 f16 weights (K = 9 taps, rest zero) ----
    // half h holds K = (h<8 ? h : h+8) + 8*hi   (ISA 7.12.2, 16-bit A 16x32)
    v16h a = {};
    {
        const int c = wave * 16 + mr;
        #pragma unroll
        for (int h2 = 0; h2 < 16; ++h2) {
            int K    = (h2 < 8 ? h2 : h2 + 8) + (hi ? 8 : 0);
            int Keff = (K < 9) ? K : 0;                 // clamp: always in-bounds
            float wv = wgt[c * 9 + Keff];
            a[h2] = (K < 9) ? (_Float16)wv : (_Float16)0.0f;
        }
    }

    int T = Tptr[0];
    __builtin_assume(T > 0);
    const float invtau = 1.0f / tauptr[0];
    const float invT   = 1.0f / (float)T;

    // ---- hoist tile-invariant per-lane channel constants & output offsets ----
    const int cbase = wave * 16 + (hi ? 8 : 0);         // D row -> channel base
    float    cinv[8], cbs[8];
    unsigned obase[8];                                  // 32-bit element offsets
    #pragma unroll
    for (int r = 0; r < 8; ++r) {
        int cc   = cbase + r;
        cinv[r]  = s_inv[cc];
        cbs[r]   = s_bias[cc];
        obase[r] = (unsigned)((((unsigned)n * C_OUT + cc) * HH + hrow) * WW + mr);
    }

    // ---- sweep 8 pixel tiles of 16 along the row ----
    for (int t = 0; t < 8; ++t) {
        const int xN = t * 16 + mr;   // pixel column for this lane (B col = D col)

        // B fragment: 32x16 f16 im2col patch.
        // half h holds K = (h<8 ? h : h+8) + 8*hi, column N = lane&15 (ISA 7.12.2 B layout)
        v16h b = {};
        #pragma unroll
        for (int h2 = 0; h2 < 16; ++h2) {
            int K    = (h2 < 8 ? h2 : h2 + 8) + (hi ? 8 : 0);
            int Keff = (K < 9) ? K : 0;                 // clamp: always in-bounds
            int dy = Keff / 3;                          // 0..2
            int dx = Keff % 3;                          // 0..2 ; srow col = x + dx
            _Float16 pv = srow[dy][xN + dx];
            b[h2] = (K < 9) ? pv : (_Float16)0.0f;
        }

        // D = A x B  (conv as GEMM, f32 accumulate)
        v8f acc = {};
        acc = __builtin_amdgcn_wmma_f32_16x16x32_f16(
            /*neg_a=*/false, a, /*neg_b=*/false, b,
            /*c_mod=*/(short)0, acc, /*reuse_a=*/false, /*reuse_b=*/false);

        // ---- BN then LIF firing rate: one dynamic-T loop, 8-wide vector body ----
        float cbn[8], vv[8], fr[8];
        #pragma unroll
        for (int r = 0; r < 8; ++r) {
            cbn[r] = acc[r] * cinv[r] + cbs[r];
            vv[r]  = 0.0f;
            fr[r]  = 0.0f;
        }
        for (int st = 0; st < T; ++st) {
            #pragma unroll
            for (int r = 0; r < 8; ++r) {
                float hcur = vv[r] + (cbn[r] - vv[r]) * invtau; // H = V + (X-V)/tau
                float s    = (hcur >= 1.0f) ? 1.0f : 0.0f;
                vv[r] = hcur - hcur * s;                        // hard reset to 0
                fr[r] += s;
            }
        }
        // ---- streaming (non-temporal) stores: write-once output > L2 ----
        #pragma unroll
        for (int r = 0; r < 8; ++r) {
            __builtin_nontemporal_store(fr[r] * invT,
                                        out + obase[r] + (unsigned)(t * 16));
        }
    }
}

extern "C" void kernel_launch(void* const* d_in, const int* in_sizes, int n_in,
                              void* d_out, int out_size, void* d_ws, size_t ws_size,
                              hipStream_t stream) {
    const float* x     = (const float*)d_in[0];
    const float* wgt   = (const float*)d_in[1];
    const float* gamma = (const float*)d_in[2];
    const float* beta  = (const float*)d_in[3];
    const float* rmean = (const float*)d_in[4];
    const float* rvar  = (const float*)d_in[5];
    const int*   Tptr  = (const int*)d_in[6];
    const float* taup  = (const float*)d_in[7];
    float* out = (float*)d_out;

    dim3 grid(HH, NB);   // one block per (output row, image)
    dim3 block(256);     // 8 waves
    ConvEnc_lif_wmma_kernel<<<grid, block, 0, stream>>>(
        x, wgt, gamma, beta, rmean, rvar, Tptr, taup, out);
}
